// VariationalRNN_76424648065252
// MI455X (gfx1250) — matrix operand order repaired and checked by reading, not compiled, not run on hardware
//
#include <hip/hip_runtime.h>
#include <hip/hip_bf16.h>

// ---------------- problem constants ----------------
constexpr int kIn  = 512;
constexpr int kHid = 256;
constexpr int kLat = 128;
constexpr int kB   = 64;
constexpr int kT   = 512;
constexpr int kNW  = 16;   // workgroups in the cluster (one WGP each)

typedef __attribute__((ext_vector_type(16))) __bf16 v16bf;
typedef __attribute__((ext_vector_type(8)))  __bf16 v8bf;
typedef __attribute__((ext_vector_type(8)))  float  v8f;

// ---------------- LDS weight arena layout (elements, 2B each) ----------------
// Row padding of 8 bf16 (16B) rotates banks by 4 dwords per row -> conflict-free
// b128 B-fragment loads.
constexpr int kP     = 8;
constexpr int oWpo1  = 0;
constexpr int oWpr1  = oWpo1 + 16 * (512 + kP);
constexpr int oWpo2  = oWpr1 + 16 * (256 + kP);
constexpr int oWpr2  = oWpo2 + 16 * (256 + kP);
constexpr int oWih0  = oWpr2 + 16 * (256 + kP);
constexpr int oWhh0  = oWih0 + 48 * (384 + kP);
constexpr int oWih1  = oWhh0 + 48 * (256 + kP);
constexpr int oWhh1  = oWih1 + 48 * (256 + kP);
constexpr int oWd1   = oWhh1 + 48 * (256 + kP);
constexpr int oWd2   = oWd1 + 16 * (384 + kP);
constexpr int kLdsWElems = oWd2 + 64 * (256 + kP);   // 100992 elems = 197.2 KB

// ---------------- WMMA helpers ----------------
__device__ __forceinline__ v8f wmma_bf16(v16bf a, v16bf b, v8f c) {
  // emits v_wmma_f32_16x16x32_bf16
  return __builtin_amdgcn_wmma_f32_16x16x32_bf16(false, a, false, b, (short)0, c,
                                                 false, false);
}

// A fragment (16x32 bf16, row-major source). Lane l: row = l&15,
// K chunks: lanes 0-15 -> K {0..15}, lanes 16-31 -> K {16..31}.
__device__ __forceinline__ v16bf load_a_bf(const __bf16* A, int lda, int m0,
                                           int k0, int lane) {
  int m    = m0 + (lane & 15);
  int koff = (lane & 16) ? 16 : 0;
  const v8bf lo = *(const v8bf*)(A + (size_t)m * lda + k0 + koff);
  const v8bf hi = *(const v8bf*)(A + (size_t)m * lda + k0 + 8 + koff);
  return __builtin_shufflevector(lo, hi, 0, 1, 2, 3, 4, 5, 6, 7, 8, 9, 10, 11,
                                 12, 13, 14, 15);
}

// B fragment (32x16) from weight W stored as [N][K] row-major (x @ W^T GEMM):
// lane l holds column n0+(l&15); K packed 2/VGPR, lanes 16-31 take K+16.
// Global-memory version (used by the parallel x_proj GEMM).
__device__ __forceinline__ v16bf load_b_bf(const __bf16* W, int ldk, int n0,
                                           int k0, int lane) {
  int n  = n0 + (lane & 15);
  int kb = k0 + ((lane & 16) ? 16 : 0);
  return *(const v16bf*)(W + (size_t)n * ldk + kb);
}

// LDS version: two 16B ds loads (row bases are 16B aligned with the pad).
__device__ __forceinline__ v16bf load_b_sm(const __bf16* W, int ldk, int n0,
                                           int k0, int lane) {
  int n  = n0 + (lane & 15);
  int kb = k0 + ((lane & 16) ? 16 : 0);
  const __bf16* p = W + n * ldk + kb;
  v8bf lo = *(const v8bf*)p;
  v8bf hi = *(const v8bf*)(p + 8);
  return __builtin_shufflevector(lo, hi, 0, 1, 2, 3, 4, 5, 6, 7, 8, 9, 10, 11,
                                 12, 13, 14, 15);
}

struct ASeg { const __bf16* p; int lda; int kn; };

// Single 16x16 output tile, B sourced from LDS weight arena.
__device__ __forceinline__ v8f tile_gemm_sm(const ASeg* segs, int nseg,
                                            const __bf16* W, int ldk, int n0,
                                            int m0, int lane, v8f acc) {
  int kg = 0;
  for (int s = 0; s < nseg; ++s) {
    for (int k0 = 0; k0 < segs[s].kn; k0 += 32) {
      acc = wmma_bf16(load_a_bf(segs[s].p, segs[s].lda, m0, k0, lane),
                      load_b_sm(W, ldk, n0, kg, lane), acc);
      kg += 32;
    }
  }
  return acc;
}

// ---------------- math ----------------
__device__ __forceinline__ float gelu_exact(float x) {
  return 0.5f * x * (1.0f + erff(x * 0.70710678118654752440f));
}
__device__ __forceinline__ float sigmoidf(float x) {
  return 1.0f / (1.0f + expf(-x));
}

// D-tile -> bias + GELU -> bf16 row-major store.
__device__ __forceinline__ void store_gelu_bf(__bf16* O, int ldo, int m0,
                                              int n0, int lane, v8f d,
                                              const float* bias) {
  int nl = lane & 15, n = n0 + nl;
  float bv = bias[n];
  int mb = m0 + ((lane & 16) ? 8 : 0);
#pragma unroll
  for (int r = 0; r < 8; ++r)
    O[(size_t)(mb + r) * ldo + n] = (__bf16)gelu_exact(d[r] + bv);
}

// ---------------- cluster barrier (release -> signal/wait -> acquire) -------
__device__ __forceinline__ void cluster_sync() {
  __threadfence();                       // make our L2 writes visible
  __syncthreads();
  __builtin_amdgcn_s_cluster_barrier();  // s_barrier_signal -3 ; s_barrier_wait -3
  __threadfence();                       // drop stale WGP$ lines
}

// Stage nRows consecutive rows (global rows gr0..) of a [N][K] bf16 weight into
// the LDS arena with K+kP padded rows. 16B chunks, whole block cooperates.
__device__ __forceinline__ void stage(const __bf16* Wg, int K, int gr0,
                                      int nRows, __bf16* dst, int tid) {
  const int cpr = K / 8;                 // 16B chunks per row
  const int chunks = nRows * cpr;
  for (int i = tid; i < chunks; i += 256) {
    int row = i / cpr, c = i - row * cpr;
    uint4 v = *(const uint4*)(Wg + (size_t)(gr0 + row) * K + c * 8);
    *(uint4*)(dst + row * (K + kP) + c * 8) = v;
  }
}

// ---------------- GRU layer (one step, one WGP owns 16 h-columns) ----------
// Gate weights live in LDS as 48 local rows: [0..15]=r, [16..31]=z, [32..47]=n.
__device__ __forceinline__ void gru_layer(int wave, int lane, int c0,
                                          const ASeg* giSegs, int nGi,
                                          const __bf16* hbPrev,
                                          const __bf16* WihL, int ldkIh,
                                          const __bf16* WhhL, int ldkHh,
                                          const float* bih, const float* bhh,
                                          const float* hprevf, float* houtf,
                                          __bf16* houtb, float* lds_gh) {
  v8f g0 = {}, g1 = {}, g2 = {};
  const int nl = lane & 15;
  if (wave < 4) {
    const int m0 = wave * 16;
    int kg = 0;
    for (int s = 0; s < nGi; ++s)
      for (int k0 = 0; k0 < giSegs[s].kn; k0 += 32) {
        v16bf a = load_a_bf(giSegs[s].p, giSegs[s].lda, m0, k0, lane);
        g0 = wmma_bf16(a, load_b_sm(WihL, ldkIh, 0, kg, lane), g0);
        g1 = wmma_bf16(a, load_b_sm(WihL, ldkIh, 16, kg, lane), g1);
        g2 = wmma_bf16(a, load_b_sm(WihL, ldkIh, 32, kg, lane), g2);
        kg += 32;
      }
  } else {
    const int mi = wave - 4, m0 = mi * 16;
    for (int k0 = 0; k0 < kHid; k0 += 32) {
      v16bf a = load_a_bf(hbPrev, kHid, m0, k0, lane);
      g0 = wmma_bf16(a, load_b_sm(WhhL, ldkHh, 0, k0, lane), g0);
      g1 = wmma_bf16(a, load_b_sm(WhhL, ldkHh, 16, k0, lane), g1);
      g2 = wmma_bf16(a, load_b_sm(WhhL, ldkHh, 32, k0, lane), g2);
    }
    const int n = c0 + nl;
    float* base = lds_gh + mi * 3 * 256;
#pragma unroll
    for (int r = 0; r < 8; ++r) {
      int row = ((lane & 16) ? 8 : 0) + r;
      base[0 * 256 + row * 16 + nl] = g0[r] + bhh[n];
      base[1 * 256 + row * 16 + nl] = g1[r] + bhh[256 + n];
      base[2 * 256 + row * 16 + nl] = g2[r] + bhh[512 + n];
    }
  }
  __syncthreads();
  if (wave < 4) {
    const int m0 = wave * 16;
    const int n = c0 + nl;
    const float b_r = bih[n], b_z = bih[256 + n], b_n = bih[512 + n];
    const float* base = lds_gh + wave * 3 * 256;
#pragma unroll
    for (int r = 0; r < 8; ++r) {
      int row = ((lane & 16) ? 8 : 0) + r;
      int m = m0 + row;
      float rg   = sigmoidf(g0[r] + b_r + base[0 * 256 + row * 16 + nl]);
      float zg   = sigmoidf(g1[r] + b_z + base[1 * 256 + row * 16 + nl]);
      float cand = tanhf(g2[r] + b_n + rg * base[2 * 256 + row * 16 + nl]);
      float hp = hprevf[m * kHid + n];
      float h = (1.0f - zg) * cand + zg * hp;
      houtf[m * kHid + n] = h;
      houtb[m * kHid + n] = (__bf16)h;
    }
  }
}

// ---------------- kernels ----------------
__global__ void cvt_bf16_kernel(const float* __restrict__ s,
                                __bf16* __restrict__ d, int n) {
  int i = blockIdx.x * blockDim.x + threadIdx.x;
  if (i < n) d[i] = (__bf16)s[i];
}

__global__ void init_h_kernel(const float* __restrict__ hidden, float* h0f,
                              float* h1f, __bf16* h0b, __bf16* h1b) {
  int i = blockIdx.x * blockDim.x + threadIdx.x;
  if (i < kB * kHid) {
    float v0 = hidden[i];
    float v1 = hidden[kB * kHid + i];
    h0f[i] = v0; h1f[i] = v1;           // parity-0 buffers at base
    h0b[i] = (__bf16)v0; h1b[i] = (__bf16)v1;
  }
}

// x_proj = x_seq @ Wp^T + bp, all B*T rows in parallel; bf16-in, bf16-out.
__global__ __launch_bounds__(256) void xproj_kernel(
    const __bf16* __restrict__ xseqb, const __bf16* __restrict__ wpb,
    const float* __restrict__ bp, __bf16* __restrict__ xproj) {
  int wave = threadIdx.x >> 5, lane = threadIdx.x & 31;
  int tile = blockIdx.x * 8 + wave;      // 2048 M-tiles x 16 N-tiles
  int m0 = (tile >> 4) * 16, n0 = (tile & 15) * 16;
  v8f acc = {};
  for (int k0 = 0; k0 < kIn; k0 += 32)
    acc = wmma_bf16(load_a_bf(xseqb, kIn, m0, k0, lane),
                    load_b_bf(wpb, kIn, n0, k0, lane), acc);
  int nl = lane & 15, n = n0 + nl;
  float bv = bp[n];
  int mb = m0 + ((lane & 16) ? 8 : 0);
#pragma unroll
  for (int r = 0; r < 8; ++r)
    xproj[(size_t)(mb + r) * kHid + n] = (__bf16)(acc[r] + bv);
}

struct VrnnArgs {
  const __bf16* xproj;                             // (B,T,HID) bf16
  const __bf16 *wih0, *whh0, *wih1, *whh1;
  const __bf16 *wpr1, *wpr2, *wpo1, *wpo2, *wd1, *wd2;
  const float *bih0, *bhh0, *bih1, *bhh1;
  const float *bpr1, *bpr2, *bpo1, *bpo2, *bd1, *bd2;
  float  *h0f, *h1f;                               // [2][B*HID]
  __bf16 *h0b, *h1b;                               // [2][B*HID]
  __bf16 *zb;                                      // [B*LAT]
  __bf16 *a1b, *a2b, *d1b;                         // [B*HID]
  float  *qpf, *ppf;                               // [B*2LAT]
  float  *klpart;                                  // [kNW]
  float  *out_mu, *out_lv, *out_kl, *out_hid;
};

// Persistent recurrence: 16 workgroups (a cluster), WGP w owns column-slice
// [w*16, w*16+16) of every hidden-sized layer and 64 decoder columns. All of
// this WGP's weights are staged once into the 320KB LDS (~197KB) and every
// B-fragment of the 512-step loop is fed from LDS.
__global__ __launch_bounds__(256) void vrnn_kernel(VrnnArgs g) {
  const int w = blockIdx.x, tid = threadIdx.x;
  const int wave = tid >> 5, lane = tid & 31;
  const int c0 = w * 16;
  const int ldx = kT * kHid;

  extern __shared__ __align__(16) __bf16 smw[];    // weight arena
  __shared__ float lds_gh[4 * 3 * 256];
  __shared__ float lds_red[256];

  // ---- one-time weight staging into LDS ----
  stage(g.wpo1, 512, c0, 16, smw + oWpo1, tid);
  stage(g.wpr1, 256, c0, 16, smw + oWpr1, tid);
  stage(g.wpo2, 256, c0, 16, smw + oWpo2, tid);
  stage(g.wpr2, 256, c0, 16, smw + oWpr2, tid);
  stage(g.wih0, 384, c0,       16, smw + oWih0 +  0 * (384 + kP), tid);
  stage(g.wih0, 384, 256 + c0, 16, smw + oWih0 + 16 * (384 + kP), tid);
  stage(g.wih0, 384, 512 + c0, 16, smw + oWih0 + 32 * (384 + kP), tid);
  stage(g.whh0, 256, c0,       16, smw + oWhh0 +  0 * (256 + kP), tid);
  stage(g.whh0, 256, 256 + c0, 16, smw + oWhh0 + 16 * (256 + kP), tid);
  stage(g.whh0, 256, 512 + c0, 16, smw + oWhh0 + 32 * (256 + kP), tid);
  stage(g.wih1, 256, c0,       16, smw + oWih1 +  0 * (256 + kP), tid);
  stage(g.wih1, 256, 256 + c0, 16, smw + oWih1 + 16 * (256 + kP), tid);
  stage(g.wih1, 256, 512 + c0, 16, smw + oWih1 + 32 * (256 + kP), tid);
  stage(g.whh1, 256, c0,       16, smw + oWhh1 +  0 * (256 + kP), tid);
  stage(g.whh1, 256, 256 + c0, 16, smw + oWhh1 + 16 * (256 + kP), tid);
  stage(g.whh1, 256, 512 + c0, 16, smw + oWhh1 + 32 * (256 + kP), tid);
  stage(g.wd1, 384, c0, 16, smw + oWd1, tid);
  stage(g.wd2, 256, w * 64, 64, smw + oWd2, tid);
  __syncthreads();

  float kl_acc = 0.0f;

  for (int t = 0; t < kT; ++t) {
    const int p = t & 1, np = p ^ 1;
    const __bf16* xpt   = g.xproj + (size_t)t * kHid;   // row stride = T*HID
    const __bf16* h1b_p = g.h1b + p * (kB * kHid);
    const __bf16* h0b_p = g.h0b + p * (kB * kHid);
    const float*  h0f_p = g.h0f + p * (kB * kHid);
    const float*  h1f_p = g.h1f + p * (kB * kHid);
    float*  h0f_n = g.h0f + np * (kB * kHid);
    float*  h1f_n = g.h1f + np * (kB * kHid);
    __bf16* h0b_n = g.h0b + np * (kB * kHid);
    __bf16* h1b_n = g.h1b + np * (kB * kHid);

    // Phase A: a1 = gelu(cat(h1_prev, xt) @ Wpo1^T + bpo1) (waves 0-3)
    //          a2 = gelu(h1_prev @ Wpr1^T + bpr1)          (waves 4-7)
    if (wave < 4) {
      int m0 = wave * 16;
      ASeg segs[2] = { { h1b_p, kHid, kHid }, { xpt, ldx, kHid } };
      v8f acc = {};
      acc = tile_gemm_sm(segs, 2, smw + oWpo1, 512 + kP, 0, m0, lane, acc);
      store_gelu_bf(g.a1b, kHid, m0, c0, lane, acc, g.bpo1);
    } else {
      int m0 = (wave - 4) * 16;
      ASeg segs[1] = { { h1b_p, kHid, kHid } };
      v8f acc = {};
      acc = tile_gemm_sm(segs, 1, smw + oWpr1, 256 + kP, 0, m0, lane, acc);
      store_gelu_bf(g.a2b, kHid, m0, c0, lane, acc, g.bpr1);
    }
    cluster_sync();

    // Phase B: qp = a1 @ Wpo2^T + bpo2 (waves 0-3), pp = a2 @ Wpr2^T (4-7);
    // z = q_mu (first 128 cols of qp) stored bf16.
    {
      bool isq = wave < 4;
      int m0 = (isq ? wave : wave - 4) * 16;
      ASeg segs[1] = { { isq ? g.a1b : g.a2b, kHid, kHid } };
      v8f acc = {};
      acc = tile_gemm_sm(segs, 1, smw + (isq ? oWpo2 : oWpr2), 256 + kP, 0, m0,
                         lane, acc);
      int nl = lane & 15, n = c0 + nl;
      float bv = (isq ? g.bpo2 : g.bpr2)[n];
      int mb = m0 + ((lane & 16) ? 8 : 0);
      float* outp = isq ? g.qpf : g.ppf;
#pragma unroll
      for (int r = 0; r < 8; ++r) {
        float v = acc[r] + bv;
        outp[(mb + r) * (2 * kLat) + n] = v;
        if (isq && n < kLat) g.zb[(mb + r) * kLat + n] = (__bf16)v;
      }
    }
    cluster_sync();

    // Phase C: KL accumulation (all threads) + GRU layer 0.
    {
      int base = w * 512 + tid * 2;
#pragma unroll
      for (int u = 0; u < 2; ++u) {
        int idx = base + u;
        int b = idx >> 7, l = idx & (kLat - 1);
        float pm  = g.ppf[b * 2 * kLat + l];
        float plv = g.ppf[b * 2 * kLat + kLat + l];
        float qm  = g.qpf[b * 2 * kLat + l];
        float qlv = g.qpf[b * 2 * kLat + kLat + l];
        float dm = qm - pm;
        kl_acc += 0.5f * (plv - qlv + (expf(qlv) + dm * dm) * expf(-plv) - 1.0f);
      }
    }
    {
      ASeg gi0[2] = { { xpt, ldx, kHid }, { g.zb, kLat, kLat } };
      gru_layer(wave, lane, c0, gi0, 2, h0b_p, smw + oWih0, 384 + kP,
                smw + oWhh0, 256 + kP, g.bih0, g.bhh0, h0f_p, h0f_n, h0b_n,
                lds_gh);
    }
    cluster_sync();

    // Phase D: GRU layer 1 (input = new h0).
    {
      ASeg gi1[1] = { { h0b_n, kHid, kHid } };
      gru_layer(wave, lane, c0, gi1, 1, h1b_p, smw + oWih1, 256 + kP,
                smw + oWhh1, 256 + kP, g.bih1, g.bhh1, h1f_p, h1f_n, h1b_n,
                lds_gh);
    }
    cluster_sync();

    // Phase E: d1 = gelu(cat(h1_new, z) @ Wd1^T + bd1)
    if (wave < 4) {
      int m0 = wave * 16;
      ASeg segs[2] = { { h1b_n, kHid, kHid }, { g.zb, kLat, kLat } };
      v8f acc = {};
      acc = tile_gemm_sm(segs, 2, smw + oWd1, 384 + kP, 0, m0, lane, acc);
      store_gelu_bf(g.d1b, kHid, m0, c0, lane, acc, g.bd1);
    }
    cluster_sync();

    // Phase F: op = d1 @ Wd2^T + bd2 -> pred_mu / pred_lv (64 cols per WGP).
#pragma unroll
    for (int i = 0; i < 2; ++i) {
      int tIdx = wave * 2 + i;
      int m0 = (tIdx & 3) * 16;
      int n0loc = (tIdx >> 2) * 16;      // local row in the wd2 LDS slice
      ASeg segs[1] = { { g.d1b, kHid, kHid } };
      v8f acc = {};
      acc = tile_gemm_sm(segs, 1, smw + oWd2, 256 + kP, n0loc, m0, lane, acc);
      int nl = lane & 15, n = w * 64 + n0loc + nl;
      float bv = g.bd2[n];
      int mb = m0 + ((lane & 16) ? 8 : 0);
      if (n < kIn) {
#pragma unroll
        for (int r = 0; r < 8; ++r)
          g.out_mu[(size_t)(mb + r) * (kT * kIn) + (size_t)t * kIn + n] =
              acc[r] + bv;
      } else {
#pragma unroll
        for (int r = 0; r < 8; ++r)
          g.out_lv[(size_t)(mb + r) * (kT * kIn) + (size_t)t * kIn + (n - kIn)] =
              acc[r] + bv;
      }
    }
    // No barrier needed here: next-iteration phases re-synchronize before any
    // buffer written after this point is consumed.
  }

  // Deterministic KL reduction: thread -> block (ordered) -> cluster (ordered).
  lds_red[tid] = kl_acc;
  __syncthreads();
  if (tid == 0) {
    float s = 0.0f;
    for (int i = 0; i < 256; ++i) s += lds_red[i];
    g.klpart[w] = s;
  }
  cluster_sync();
  if (w == 0 && tid == 0) {
    float s = 0.0f;
    for (int i = 0; i < kNW; ++i) s += g.klpart[i];
    g.out_kl[0] = s / (float)(kB * kT);
  }
  // Final hidden state (T even -> parity 0 buffers hold the last write).
  for (int idx = w * 256 + tid; idx < 2 * kB * kHid; idx += kNW * 256) {
    int layer = idx / (kB * kHid);
    int o = idx - layer * (kB * kHid);
    g.out_hid[idx] = (layer == 0) ? g.h0f[o] : g.h1f[o];
  }
}

// ---------------- host launcher ----------------
extern "C" void kernel_launch(void* const* d_in, const int* in_sizes, int n_in,
                              void* d_out, int out_size, void* d_ws,
                              size_t ws_size, hipStream_t stream) {
  (void)in_sizes; (void)n_in; (void)out_size; (void)ws_size;

  const float* x_seq  = (const float*)d_in[0];
  const float* hidden = (const float*)d_in[1];
  const float* Wp   = (const float*)d_in[2];  const float* bp   = (const float*)d_in[3];
  const float* Wih0 = (const float*)d_in[4];  const float* Whh0 = (const float*)d_in[5];
  const float* bih0 = (const float*)d_in[6];  const float* bhh0 = (const float*)d_in[7];
  const float* Wih1 = (const float*)d_in[8];  const float* Whh1 = (const float*)d_in[9];
  const float* bih1 = (const float*)d_in[10]; const float* bhh1 = (const float*)d_in[11];
  const float* Wpr1 = (const float*)d_in[12]; const float* bpr1 = (const float*)d_in[13];
  const float* Wpr2 = (const float*)d_in[14]; const float* bpr2 = (const float*)d_in[15];
  const float* Wpo1 = (const float*)d_in[16]; const float* bpo1 = (const float*)d_in[17];
  const float* Wpo2 = (const float*)d_in[18]; const float* bpo2 = (const float*)d_in[19];
  const float* Wd1  = (const float*)d_in[20]; const float* bd1  = (const float*)d_in[21];
  const float* Wd2  = (const float*)d_in[22]; const float* bd2  = (const float*)d_in[23];

  char* wsb = (char*)d_ws;
  size_t off = 0;
  auto take = [&](size_t bytes) -> char* {
    char* ptr = wsb + off;
    off += (bytes + 255) & ~size_t(255);
    return ptr;
  };

  __bf16* xseqb  = (__bf16*)take((size_t)kB * kT * kIn * 2);
  __bf16* xprojb = (__bf16*)take((size_t)kB * kT * kHid * 2);
  __bf16* wpb    = (__bf16*)take((size_t)256 * 512 * 2);
  __bf16* wih0b  = (__bf16*)take((size_t)768 * 384 * 2);
  __bf16* whh0b  = (__bf16*)take((size_t)768 * 256 * 2);
  __bf16* wih1b  = (__bf16*)take((size_t)768 * 256 * 2);
  __bf16* whh1b  = (__bf16*)take((size_t)768 * 256 * 2);
  __bf16* wpr1b  = (__bf16*)take((size_t)256 * 256 * 2);
  __bf16* wpr2b  = (__bf16*)take((size_t)256 * 256 * 2);
  __bf16* wpo1b  = (__bf16*)take((size_t)256 * 512 * 2);
  __bf16* wpo2b  = (__bf16*)take((size_t)256 * 256 * 2);
  __bf16* wd1b   = (__bf16*)take((size_t)256 * 384 * 2);
  __bf16* wd2b   = (__bf16*)take((size_t)1024 * 256 * 2);
  float*  h0f    = (float*)take((size_t)2 * kB * kHid * 4);
  float*  h1f    = (float*)take((size_t)2 * kB * kHid * 4);
  __bf16* h0b    = (__bf16*)take((size_t)2 * kB * kHid * 2);
  __bf16* h1b    = (__bf16*)take((size_t)2 * kB * kHid * 2);
  __bf16* zb     = (__bf16*)take((size_t)kB * kLat * 2);
  __bf16* a1b    = (__bf16*)take((size_t)kB * kHid * 2);
  __bf16* a2b    = (__bf16*)take((size_t)kB * kHid * 2);
  __bf16* d1b    = (__bf16*)take((size_t)kB * kHid * 2);
  float*  qpf    = (float*)take((size_t)kB * 2 * kLat * 4);
  float*  ppf    = (float*)take((size_t)kB * 2 * kLat * 4);
  float*  klpart = (float*)take((size_t)kNW * 4);

  // 1) fp32 -> bf16 conversions (weights once, x_seq once - no redundant cvt
  //    inside the GEMM inner loops)
  struct Job { const float* s; __bf16* d; int n; };
  Job jobs[12] = {
      { x_seq, xseqb, kB * kT * kIn },
      { Wp, wpb, 256 * 512 },     { Wih0, wih0b, 768 * 384 },
      { Whh0, whh0b, 768 * 256 }, { Wih1, wih1b, 768 * 256 },
      { Whh1, whh1b, 768 * 256 }, { Wpr1, wpr1b, 256 * 256 },
      { Wpr2, wpr2b, 256 * 256 }, { Wpo1, wpo1b, 256 * 512 },
      { Wpo2, wpo2b, 256 * 256 }, { Wd1, wd1b, 256 * 384 },
      { Wd2, wd2b, 1024 * 256 } };
  for (int j = 0; j < 12; ++j)
    cvt_bf16_kernel<<<(jobs[j].n + 255) / 256, 256, 0, stream>>>(
        jobs[j].s, jobs[j].d, jobs[j].n);

  // 2) initial hidden state (fp32 + bf16 copies, parity 0)
  init_h_kernel<<<(kB * kHid + 255) / 256, 256, 0, stream>>>(hidden, h0f, h1f,
                                                             h0b, h1b);

  // 3) big parallel GEMM: x_proj for all (b,t)
  xproj_kernel<<<(kB * kT / 16) * (kHid / 16) / 8, 256, 0, stream>>>(
      xseqb, wpb, bp, xprojb);

  // 4) persistent cluster recurrence with LDS-resident weights (~197KB/WGP)
  float* outf = (float*)d_out;
  VrnnArgs a;
  a.xproj = xprojb;
  a.wih0 = wih0b; a.whh0 = whh0b; a.wih1 = wih1b; a.whh1 = whh1b;
  a.wpr1 = wpr1b; a.wpr2 = wpr2b; a.wpo1 = wpo1b; a.wpo2 = wpo2b;
  a.wd1 = wd1b;   a.wd2 = wd2b;
  a.bih0 = bih0; a.bhh0 = bhh0; a.bih1 = bih1; a.bhh1 = bhh1;
  a.bpr1 = bpr1; a.bpr2 = bpr2; a.bpo1 = bpo1; a.bpo2 = bpo2;
  a.bd1 = bd1;   a.bd2 = bd2;
  a.h0f = h0f; a.h1f = h1f; a.h0b = h0b; a.h1b = h1b;
  a.zb = zb; a.a1b = a1b; a.a2b = a2b; a.d1b = d1b;
  a.qpf = qpf; a.ppf = ppf; a.klpart = klpart;
  a.out_mu = outf;
  a.out_lv = outf + (size_t)kB * kT * kIn;
  a.out_kl = outf + (size_t)2 * kB * kT * kIn;
  a.out_hid = a.out_kl + 1;
  vrnn_kernel<<<kNW, 256, (size_t)kLdsWElems * sizeof(__bf16), stream>>>(a);
}
